// MultiHeadSelfAttention_56521769615862
// MI455X (gfx1250) — matrix operand order repaired and checked by reading, not compiled, not run on hardware
//
#include <hip/hip_runtime.h>
#include <hip/hip_fp16.h>

typedef __attribute__((ext_vector_type(16))) _Float16 v16h;
typedef __attribute__((ext_vector_type(8)))  float    v8f;
typedef __attribute__((ext_vector_type(4)))  float    v4f;

#define NB   8
#define NS   2048
#define ND   384
#define NH   6
#define NHD  64
#define MTOT (NB*NS)          // 16384 tokens

#define WMMA_F16(a,b,c) __builtin_amdgcn_wmma_f32_16x16x32_f16(false,(a),false,(b),(short)0,(c),false,false)

// ---- WMMA operand loaders -------------------------------------------------
// A operand (16x32, f16) from a row-major [rows x ld] f16 matrix.
// Lane L: row = row0 + (L&15); lanes 0-15 carry K {0..15}, lanes 16-31 carry K {16..31}.
static __device__ __forceinline__ v16h wmma_a_row(const _Float16* base, int ld, int row0, int k0, int lane) {
  const int m = lane & 15, hi = lane >> 4;
  const _Float16* p = base + (size_t)(row0 + m) * ld + k0 + hi * 16;
  v16h a;
  ((v4f*)&a)[0] = *(const v4f*)(p);        // K = k0 + hi*16 .. +7
  ((v4f*)&a)[1] = *(const v4f*)(p + 8);    // K = k0 + hi*16 + 8 .. +15
  return a;
}

// B operand (32x16, f16) from pre-swizzled packed weights: tile = 512 halves,
// lane n<16 holds column n K=0..15, lane n+16 holds column n K=16..31 (contiguous 32B per lane).
static __device__ __forceinline__ v16h wmma_b_packed(const _Float16* pack, int tile, int lane) {
  const _Float16* p = pack + (size_t)tile * 512 + lane * 16;
  v16h b;
  ((v4f*)&b)[0] = *(const v4f*)(p);
  ((v4f*)&b)[1] = *(const v4f*)(p + 8);
  return b;
}

// B operand from 16 contiguous halves per lane (32B): cols = n, K split by lane half.
static __device__ __forceinline__ v16h wmma_b_contig(const _Float16* p) {
  v16h b;
  ((v4f*)&b)[0] = *(const v4f*)(p);
  ((v4f*)&b)[1] = *(const v4f*)(p + 8);
  return b;
}

// ---- 1. LayerNorm -> f16 --------------------------------------------------
__global__ __launch_bounds__(256) void ln_kernel(const float* __restrict__ x,
                                                 const float* __restrict__ g,
                                                 const float* __restrict__ bta,
                                                 _Float16* __restrict__ Y) {
  const int lane = threadIdx.x & 31;
  const int wv   = threadIdx.x >> 5;
  const size_t row = (size_t)blockIdx.x * 8 + wv;
  const float* xr = x + row * ND;
  float v[12];
  float s = 0.f;
  #pragma unroll
  for (int j = 0; j < 12; ++j) { v[j] = xr[lane + 32 * j]; s += v[j]; }
  #pragma unroll
  for (int off = 16; off >= 1; off >>= 1) s += __shfl_xor(s, off, 32);
  const float mu = s * (1.0f / ND);
  float var = 0.f;
  #pragma unroll
  for (int j = 0; j < 12; ++j) { const float d = v[j] - mu; var += d * d; }
  #pragma unroll
  for (int off = 16; off >= 1; off >>= 1) var += __shfl_xor(var, off, 32);
  const float inv = rsqrtf(var * (1.0f / ND) + 1e-5f);
  #pragma unroll
  for (int j = 0; j < 12; ++j) {
    const int c = lane + 32 * j;
    Y[row * ND + c] = (_Float16)((v[j] - mu) * inv * g[c] + bta[c]);
  }
}

// ---- 2. One-time weight repack into WMMA B-operand layout -----------------
// Tiles 0..863 : QKV   ((mat*6 + h)*12 + kc)*4 + nc   (K=384 -> 12 kchunks, N=64 -> 4 nchunks)
// Tiles 864+   : Wp    kc*24 + nc                     (K=384, N=384 -> 24 nchunks)
__global__ __launch_bounds__(32) void pack_weights(const float* __restrict__ Wq,
                                                   const float* __restrict__ Wk,
                                                   const float* __restrict__ Wv,
                                                   const float* __restrict__ Wp,
                                                   _Float16* __restrict__ qkvp,
                                                   _Float16* __restrict__ wpp) {
  const int t = blockIdx.x, lane = threadIdx.x;
  const float* src; int ldn; _Float16* dst; int kb, n;
  if (t < 864) {
    const int mat = t / 288, rem = t % 288, h = rem / 48, r2 = rem % 48, kc = r2 / 4, nc = r2 % 4;
    src = (mat == 0 ? Wq : mat == 1 ? Wk : Wv) + (size_t)h * ND * NHD;
    ldn = NHD; dst = qkvp + (size_t)t * 512;
    kb = kc * 32 + (lane >> 4) * 16; n = nc * 16 + (lane & 15);
  } else {
    const int t2 = t - 864, kc = t2 / 24, nc = t2 % 24;
    src = Wp; ldn = ND; dst = wpp + (size_t)t2 * 512;
    kb = kc * 32 + (lane >> 4) * 16; n = nc * 16 + (lane & 15);
  }
  #pragma unroll
  for (int j = 0; j < 16; ++j)
    dst[lane * 16 + j] = (_Float16)src[(size_t)(kb + j) * ldn + n];
}

// ---- 3. QKV projection GEMM -----------------------------------------------
// Each wave computes a 16x64 strip (full head width) = 48 WMMAs. Software-
// pipelined: k-step (kc+1) operands are fetched into fresh registers before
// the 4 WMMAs of k-step kc, so no load completes on the critical path.
__global__ __launch_bounds__(32) void qkv_gemm(const _Float16* __restrict__ Y,
                                               const _Float16* __restrict__ pack,
                                               const float* __restrict__ bq,
                                               const float* __restrict__ bk,
                                               const float* __restrict__ bv,
                                               _Float16* __restrict__ Q,
                                               _Float16* __restrict__ K,
                                               _Float16* __restrict__ Vt) {
  const int lane = threadIdx.x;
  const int mt = blockIdx.x;                 // 0..1023 (16-token tiles)
  const int y  = blockIdx.y;                 // 0..17 -> (mat, h)
  const int mat = y / NH, h = y % NH;
  const int tbase = (mat * NH + h) * 48;     // 12 kchunks * 4 nchunks
  v8f acc[4] = {{}, {}, {}, {}};

  v16h a_cur = wmma_a_row(Y, ND, mt * 16, 0, lane);
  v16h b_cur0 = wmma_b_packed(pack, tbase + 0, lane);
  v16h b_cur1 = wmma_b_packed(pack, tbase + 1, lane);
  v16h b_cur2 = wmma_b_packed(pack, tbase + 2, lane);
  v16h b_cur3 = wmma_b_packed(pack, tbase + 3, lane);

  #pragma unroll
  for (int kc = 0; kc < 12; ++kc) {
    v16h a_nxt = a_cur, b_nxt0 = b_cur0, b_nxt1 = b_cur1, b_nxt2 = b_cur2, b_nxt3 = b_cur3;
    if (kc + 1 < 12) {
      a_nxt  = wmma_a_row(Y, ND, mt * 16, (kc + 1) * 32, lane);
      b_nxt0 = wmma_b_packed(pack, tbase + (kc + 1) * 4 + 0, lane);
      b_nxt1 = wmma_b_packed(pack, tbase + (kc + 1) * 4 + 1, lane);
      b_nxt2 = wmma_b_packed(pack, tbase + (kc + 1) * 4 + 2, lane);
      b_nxt3 = wmma_b_packed(pack, tbase + (kc + 1) * 4 + 3, lane);
    }
    acc[0] = WMMA_F16(a_cur, b_cur0, acc[0]);
    acc[1] = WMMA_F16(a_cur, b_cur1, acc[1]);
    acc[2] = WMMA_F16(a_cur, b_cur2, acc[2]);
    acc[3] = WMMA_F16(a_cur, b_cur3, acc[3]);
    a_cur = a_nxt; b_cur0 = b_nxt0; b_cur1 = b_nxt1; b_cur2 = b_nxt2; b_cur3 = b_nxt3;
  }

  const float* bias = (mat == 0 ? bq : mat == 1 ? bk : bv) + h * NHD;
  const int n = lane & 15, hi = lane >> 4;
  if (mat < 2) {                                     // Q, K: [b,h,s,hd]
    _Float16* Out = (mat == 0 ? Q : K);
    #pragma unroll
    for (int nc = 0; nc < 4; ++nc) {
      const float bn = bias[nc * 16 + n];
      #pragma unroll
      for (int r = 0; r < 8; ++r) {
        const int tok = mt * 16 + r + 8 * hi;        // S multiple of 16 -> same batch
        const int bb = tok / NS, s = tok % NS;
        Out[(((size_t)bb * NH + h) * NS + s) * NHD + nc * 16 + n] = (_Float16)(acc[nc][r] + bn);
      }
    }
  } else {                                           // V transposed: [b,h,hd,s]
    const int tok0 = mt * 16 + 8 * hi;
    const int bb = tok0 / NS, s0 = tok0 % NS;        // 8 consecutive tokens per lane
    #pragma unroll
    for (int nc = 0; nc < 4; ++nc) {
      const float bn = bias[nc * 16 + n];
      __align__(16) _Float16 tmp[8];
      #pragma unroll
      for (int r = 0; r < 8; ++r) tmp[r] = (_Float16)(acc[nc][r] + bn);
      *(v4f*)(Vt + (((size_t)bb * NH + h) * NHD + nc * 16 + n) * NS + s0) = *(v4f*)tmp;
    }
  }
}

// ---- 4. Flash attention (unscaled scores, per reference) ------------------
// 8 waves/block share (b,h); K chunks stream into double-buffered LDS via
// CDNA5 async loads (ASYNCcnt); V read straight from transposed global layout.
__global__ __launch_bounds__(256) void attn_kernel(const _Float16* __restrict__ Qg,
                                                   const _Float16* __restrict__ Kg,
                                                   const _Float16* __restrict__ Vtg,
                                                   _Float16* __restrict__ Ctx) {
  __shared__ __align__(16) _Float16 Ks[2][32][72];  // double-buffered [key][hd], padded
  __shared__ __align__(16) _Float16 Ps[8][16][40];  // per-wave P tile 16x32, padded

  const int tid = threadIdx.x, lane = tid & 31, wv = tid >> 5;
  const int bh = blockIdx.x, bb = bh / NH, h = bh % NH;
  const int q0 = blockIdx.y * 128 + wv * 16;
  const size_t base = (((size_t)bb * NH + h) * NS) * NHD;
  const _Float16* Qb  = Qg + base;
  const _Float16* Kb  = Kg + base;
  const _Float16* Vtb = Vtg + base;                  // [hd][s], same flat base

  const v16h aq0 = wmma_a_row(Qb, NHD, q0, 0, lane);   // Q tile, hd 0..31
  const v16h aq1 = wmma_a_row(Qb, NHD, q0, 32, lane);  // Q tile, hd 32..63

  v8f o0 = {}, o1 = {}, o2 = {}, o3 = {};            // O tile 16x64 (4 C tiles)
  float mrow[8], lrow[8];
  #pragma unroll
  for (int r = 0; r < 8; ++r) { mrow[r] = -1e30f; lrow[r] = 0.f; }
  const int n = lane & 15, hi = lane >> 4;
  const int key = tid >> 3, hd0 = (tid & 7) * 8;     // staging coords: 16B/thread

  // Prologue: async-stage chunk 0 into buffer 0.
  {
    const unsigned lds = (unsigned)(size_t)&Ks[0][key][hd0];
    const unsigned long long g = (unsigned long long)(size_t)(Kb + (size_t)key * NHD + hd0);
    asm volatile("global_load_async_to_lds_b128 %0, %1, off" :: "v"(lds), "v"(g) : "memory");
    asm volatile("s_wait_asynccnt 0x0" ::: "memory");
  }
  __syncthreads();

  for (int kc = 0; kc < NS / 32; ++kc) {
    const int cur = kc & 1;
    // Kick off async stage of next chunk into the other buffer.
    if (kc + 1 < NS / 32) {
      const unsigned lds = (unsigned)(size_t)&Ks[cur ^ 1][key][hd0];
      const unsigned long long g =
          (unsigned long long)(size_t)(Kb + (size_t)((kc + 1) * 32 + key) * NHD + hd0);
      asm volatile("global_load_async_to_lds_b128 %0, %1, off" :: "v"(lds), "v"(g) : "memory");
      __builtin_prefetch(Vtb + (size_t)n * NS + (kc + 1) * 32, 0, 1);  // warm V path
    }

    // Scores: S = Q * K^T, 16 queries x 32 keys (2 C tiles, 2 k-steps each)
    v8f s0 = {}, s1 = {};
    {
      const v16h b00 = wmma_b_contig(&Ks[cur][n][hi * 16]);
      const v16h b01 = wmma_b_contig(&Ks[cur][n][32 + hi * 16]);
      const v16h b10 = wmma_b_contig(&Ks[cur][16 + n][hi * 16]);
      const v16h b11 = wmma_b_contig(&Ks[cur][16 + n][32 + hi * 16]);
      s0 = WMMA_F16(aq0, b00, s0); s0 = WMMA_F16(aq1, b01, s0);
      s1 = WMMA_F16(aq0, b10, s1); s1 = WMMA_F16(aq1, b11, s1);
    }

    // Streaming softmax. Row (r + 8*hi) lives in VGPR r across the 16 lanes of this half.
    #pragma unroll
    for (int r = 0; r < 8; ++r) {
      float mx = fmaxf(s0[r], s1[r]);
      mx = fmaxf(mx, __shfl_xor(mx, 1, 32));
      mx = fmaxf(mx, __shfl_xor(mx, 2, 32));
      mx = fmaxf(mx, __shfl_xor(mx, 4, 32));
      mx = fmaxf(mx, __shfl_xor(mx, 8, 32));
      const float mn = fmaxf(mrow[r], mx);
      const float alpha = __expf(mrow[r] - mn);
      mrow[r] = mn;
      const float p0 = __expf(s0[r] - mn);
      const float p1 = __expf(s1[r] - mn);
      float ps = p0 + p1;
      ps += __shfl_xor(ps, 1, 32);
      ps += __shfl_xor(ps, 2, 32);
      ps += __shfl_xor(ps, 4, 32);
      ps += __shfl_xor(ps, 8, 32);
      lrow[r] = lrow[r] * alpha + ps;
      o0[r] *= alpha; o1[r] *= alpha; o2[r] *= alpha; o3[r] *= alpha;
      const int row = r + 8 * hi;
      Ps[wv][row][n]      = (_Float16)p0;   // C-layout -> per-wave LDS scratch
      Ps[wv][row][16 + n] = (_Float16)p1;
    }

    // Read P back as A operand (DS ops are in-order within the wave).
    v16h ap;
    {
      const _Float16* pp = &Ps[wv][lane & 15][hi * 16];
      ((v4f*)&ap)[0] = *(const v4f*)pp;
      ((v4f*)&ap)[1] = *(const v4f*)(pp + 8);
    }

    // O += P * V: B operands straight from transposed V, 32B contiguous per lane.
    {
      const _Float16* vp = Vtb + (size_t)n * NS + kc * 32 + hi * 16;
      const v16h w0 = wmma_b_contig(vp);
      const v16h w1 = wmma_b_contig(vp + (size_t)16 * NS);
      const v16h w2 = wmma_b_contig(vp + (size_t)32 * NS);
      const v16h w3 = wmma_b_contig(vp + (size_t)48 * NS);
      o0 = WMMA_F16(ap, w0, o0);
      o1 = WMMA_F16(ap, w1, o1);
      o2 = WMMA_F16(ap, w2, o2);
      o3 = WMMA_F16(ap, w3, o3);
    }

    if (kc + 1 < NS / 32)
      asm volatile("s_wait_asynccnt 0x0" ::: "memory");   // next buffer resident
    __syncthreads();
  }

  // Normalize and write context row-major [token][h*64 + hd] as f16.
  #pragma unroll
  for (int r = 0; r < 8; ++r) {
    const float inv = 1.0f / lrow[r];
    const size_t tok = (size_t)bb * NS + q0 + r + 8 * hi;
    _Float16* dst = Ctx + tok * ND + h * NHD + n;
    dst[0]  = (_Float16)(o0[r] * inv);
    dst[16] = (_Float16)(o1[r] * inv);
    dst[32] = (_Float16)(o2[r] * inv);
    dst[48] = (_Float16)(o3[r] * inv);
  }
}

// ---- 5. Output projection + exact GELU ------------------------------------
// Each wave computes a 16x64 strip (4 N tiles), software-pipelined like qkv_gemm.
__global__ __launch_bounds__(32) void proj_gelu(const _Float16* __restrict__ Ctx,
                                                const _Float16* __restrict__ pack,
                                                const float* __restrict__ bp,
                                                float* __restrict__ out) {
  const int lane = threadIdx.x;
  const int mt = blockIdx.x, ng = blockIdx.y;        // ng: 0..5 (groups of 4 n-tiles)
  v8f acc[4] = {{}, {}, {}, {}};

  v16h a_cur = wmma_a_row(Ctx, ND, mt * 16, 0, lane);
  v16h b_cur0 = wmma_b_packed(pack, ng * 4 + 0, lane);
  v16h b_cur1 = wmma_b_packed(pack, ng * 4 + 1, lane);
  v16h b_cur2 = wmma_b_packed(pack, ng * 4 + 2, lane);
  v16h b_cur3 = wmma_b_packed(pack, ng * 4 + 3, lane);

  #pragma unroll
  for (int kc = 0; kc < 12; ++kc) {
    v16h a_nxt = a_cur, b_nxt0 = b_cur0, b_nxt1 = b_cur1, b_nxt2 = b_cur2, b_nxt3 = b_cur3;
    if (kc + 1 < 12) {
      a_nxt  = wmma_a_row(Ctx, ND, mt * 16, (kc + 1) * 32, lane);
      b_nxt0 = wmma_b_packed(pack, (kc + 1) * 24 + ng * 4 + 0, lane);
      b_nxt1 = wmma_b_packed(pack, (kc + 1) * 24 + ng * 4 + 1, lane);
      b_nxt2 = wmma_b_packed(pack, (kc + 1) * 24 + ng * 4 + 2, lane);
      b_nxt3 = wmma_b_packed(pack, (kc + 1) * 24 + ng * 4 + 3, lane);
    }
    acc[0] = WMMA_F16(a_cur, b_cur0, acc[0]);
    acc[1] = WMMA_F16(a_cur, b_cur1, acc[1]);
    acc[2] = WMMA_F16(a_cur, b_cur2, acc[2]);
    acc[3] = WMMA_F16(a_cur, b_cur3, acc[3]);
    a_cur = a_nxt; b_cur0 = b_nxt0; b_cur1 = b_nxt1; b_cur2 = b_nxt2; b_cur3 = b_nxt3;
  }

  const int n = lane & 15, hi = lane >> 4;
  #pragma unroll
  for (int i = 0; i < 4; ++i) {
    const int nc = ng * 4 + i;
    const float bn = bp[nc * 16 + n];
    #pragma unroll
    for (int r = 0; r < 8; ++r) {
      const float xv = acc[i][r] + bn;
      out[(size_t)(mt * 16 + r + 8 * hi) * ND + nc * 16 + n] =
          0.5f * xv * (1.0f + erff(xv * 0.70710678118654752f));
    }
  }
}

// ---- launch ---------------------------------------------------------------
extern "C" void kernel_launch(void* const* d_in, const int* in_sizes, int n_in,
                              void* d_out, int out_size, void* d_ws, size_t ws_size,
                              hipStream_t stream) {
  const float* x    = (const float*)d_in[0];
  const float* ln_g = (const float*)d_in[1];
  const float* ln_b = (const float*)d_in[2];
  const float* Wq   = (const float*)d_in[3];
  const float* bq   = (const float*)d_in[4];
  const float* Wk   = (const float*)d_in[5];
  const float* bk   = (const float*)d_in[6];
  const float* Wv   = (const float*)d_in[7];
  const float* bv   = (const float*)d_in[8];
  const float* Wp   = (const float*)d_in[9];
  const float* bp   = (const float*)d_in[10];
  float* out = (float*)d_out;

  char* ws = (char*)d_ws;
  const size_t SZ_ACT = (size_t)MTOT * ND * sizeof(_Float16);   // 12.58 MB each
  _Float16* Y    = (_Float16*)(ws);
  _Float16* Qh   = (_Float16*)(ws + 1 * SZ_ACT);
  _Float16* Kh   = (_Float16*)(ws + 2 * SZ_ACT);
  _Float16* Vth  = (_Float16*)(ws + 3 * SZ_ACT);   // V transposed [b,h,hd,s]
  _Float16* Ctx  = (_Float16*)(ws + 4 * SZ_ACT);
  _Float16* qkvp = (_Float16*)(ws + 5 * SZ_ACT);
  _Float16* wpp  = (_Float16*)(ws + 5 * SZ_ACT + (size_t)864 * 512 * sizeof(_Float16));
  // total ~64.1 MB of d_ws

  hipLaunchKernelGGL(ln_kernel,    dim3(MTOT / 8),          dim3(256), 0, stream, x, ln_g, ln_b, Y);
  hipLaunchKernelGGL(pack_weights, dim3(1152),              dim3(32),  0, stream, Wq, Wk, Wv, Wp, qkvp, wpp);
  hipLaunchKernelGGL(qkv_gemm,     dim3(MTOT / 16, 18),     dim3(32),  0, stream, Y, qkvp, bq, bk, bv, Qh, Kh, Vth);
  hipLaunchKernelGGL(attn_kernel,  dim3(NB * NH, NS / 128), dim3(256), 0, stream, Qh, Kh, Vth, Ctx);
  hipLaunchKernelGGL(proj_gelu,    dim3(MTOT / 16, 6),      dim3(32),  0, stream, Ctx, wpp, bp, out);
}
